// MultiHeadAttention_12378095747408
// MI455X (gfx1250) — matrix operand order repaired and checked by reading, not compiled, not run on hardware
//
#include <hip/hip_runtime.h>

// ---------------------------------------------------------------------------
// MI455X (gfx1250) multi-head attention forward.
// Pipeline: 3x projection GEMM (f32 -> bf16 WMMA, head layout)
//           -> flash attention (bf16 WMMA, online softmax, async-LDS staging)
//           -> output projection GEMM (bf16 -> f32, async-LDS staging).
// Matrix ops: v_wmma_f32_16x16x32_bf16 (wave32).
// Data movement: global_load_async_to_lds_b128 (ASYNCcnt) + global_prefetch.
// ---------------------------------------------------------------------------

typedef __attribute__((ext_vector_type(16))) __bf16 v16bf;
typedef __attribute__((ext_vector_type(8)))  float  v8f;

#define DEV static __device__ __forceinline__

constexpr int BATCH   = 4;
constexpr int SEQ     = 2048;
constexpr int DMODEL  = 2048;
constexpr int NHEADS  = 16;
constexpr int DK      = 128;
constexpr int MROWS   = BATCH * SEQ;     // 8192

DEV v8f wmma_bf16(v16bf a, v16bf b, v8f c) {
  return __builtin_amdgcn_wmma_f32_16x16x32_bf16(
      /*neg_a=*/false, a, /*neg_b=*/false, b,
      /*c_mod=*/(short)0, c, /*reuse_a=*/false, /*reuse_b=*/false);
}

// Async copy of 32 bytes (2 x b128) from global to LDS; tracked by ASYNCcnt.
// INST_OFFSET applies to both the LDS and global addresses (ISA 10.x).
DEV void async_copy_32B(unsigned lds_off, const void* gptr) {
  asm volatile(
      "global_load_async_to_lds_b128 %0, %1, off\n\t"
      "global_load_async_to_lds_b128 %0, %1, off offset:16"
      :: "v"(lds_off), "v"(gptr) : "memory");
}
DEV void wait_async0() { asm volatile("s_wait_asynccnt 0" ::: "memory"); }
DEV unsigned lds_off32(const void* p) {  // flat->LDS: offset = addr[31:0]
  return (unsigned)(unsigned long long)p;
}

// A-fragment (16x32 bf16), contiguous K (matches B):
//   lanes 0-15 : K = {0..15};  lanes 16-31 : K = {16..31}
DEV v16bf load_a_frag(const __bf16* rowptr, int lh) {
  const __bf16* p = rowptr + lh * 16;
  v16bf a;
#pragma unroll
  for (int j = 0; j < 16; ++j) a[j] = p[j];
  return a;
}

// B-fragment (32x16 bf16): lane holds col n = lane&15, K = (lane>>4)*16 + j.
DEV v16bf load_b_frag(const __bf16* p) {
  v16bf b;
#pragma unroll
  for (int j = 0; j < 16; ++j) b[j] = p[j];
  return b;
}

// ---------------------------------------------------------------------------
// GEMM: C[M,N] = A[M,K] * Bw[N,K]^T  (i.e. y = x @ W.T).
// Block tile 128x128x32, 8 waves, wave tile 64x32 (4x2 WMMA tiles).
// AT = float (convert in VGPRs) or __bf16 (async copy straight to LDS).
// ---------------------------------------------------------------------------
template <typename AT, typename OT, bool HEAD_LAYOUT>
__global__ __launch_bounds__(256) void gemm_wmma(
    const AT* __restrict__ A, const float* __restrict__ Bw,
    OT* __restrict__ C, int M, int N, int K) {
  constexpr int LDA = 40;  // 32 + pad, keeps 16B alignment (80B row)
  __shared__ __bf16 As[128 * LDA];
  __shared__ __bf16 Bs[128 * LDA];

  const int tid  = threadIdx.x;
  const int lane = tid & 31;
  const int wave = tid >> 5;
  const int l16  = lane & 15;
  const int lh   = lane >> 4;
  const int wm   = wave >> 2;  // 0..1  -> 64-row group
  const int wn   = wave & 3;   // 0..3  -> 32-col group
  const int bm   = blockIdx.y;
  const int bn   = blockIdx.x;

  v8f acc[4][2];
#pragma unroll
  for (int mi = 0; mi < 4; ++mi)
#pragma unroll
    for (int ni = 0; ni < 2; ++ni) acc[mi][ni] = (v8f)0.0f;

  const int r  = tid >> 1;          // 0..127 tile row for loads
  const int cb = (tid & 1) * 16;    // 0 or 16

  for (int k0 = 0; k0 < K; k0 += 32) {
    __syncthreads();
    bool used_async = false;
    // ---- stage A tile ----
    if constexpr (sizeof(AT) == 2) {
      async_copy_32B(lds_off32(&As[r * LDA + cb]),
                     A + (size_t)(bm * 128 + r) * K + k0 + cb);
      used_async = true;
    } else {
      union { __bf16 h[16]; uint4 u[2]; } t;
      const float4* ga = (const float4*)(A + (size_t)(bm * 128 + r) * K + k0 + cb);
#pragma unroll
      for (int q4 = 0; q4 < 4; ++q4) {
        float4 f = ga[q4];
        t.h[q4 * 4 + 0] = (__bf16)f.x; t.h[q4 * 4 + 1] = (__bf16)f.y;
        t.h[q4 * 4 + 2] = (__bf16)f.z; t.h[q4 * 4 + 3] = (__bf16)f.w;
      }
      *(uint4*)&As[r * LDA + cb]     = t.u[0];
      *(uint4*)&As[r * LDA + cb + 8] = t.u[1];
    }
    // ---- stage B tile (weights, f32 -> bf16 in VGPRs) ----
    {
      union { __bf16 h[16]; uint4 u[2]; } t;
      const float4* gb = (const float4*)(Bw + (size_t)(bn * 128 + r) * K + k0 + cb);
#pragma unroll
      for (int q4 = 0; q4 < 4; ++q4) {
        float4 f = gb[q4];
        t.h[q4 * 4 + 0] = (__bf16)f.x; t.h[q4 * 4 + 1] = (__bf16)f.y;
        t.h[q4 * 4 + 2] = (__bf16)f.z; t.h[q4 * 4 + 3] = (__bf16)f.w;
      }
      *(uint4*)&Bs[r * LDA + cb]     = t.u[0];
      *(uint4*)&Bs[r * LDA + cb + 8] = t.u[1];
    }
    // ---- prefetch next K-step into L2/WGP$ ----
    if (k0 + 32 < K) {
      __builtin_prefetch(A  + (size_t)(bm * 128 + r) * K + k0 + 32 + cb);
      __builtin_prefetch(Bw + (size_t)(bn * 128 + r) * K + k0 + 32 + cb);
    }
    if (used_async) wait_async0();
    __syncthreads();

    v16bf afr[4], bfr[2];
#pragma unroll
    for (int mi = 0; mi < 4; ++mi)
      afr[mi] = load_a_frag(&As[(wm * 64 + mi * 16 + l16) * LDA], lh);
#pragma unroll
    for (int ni = 0; ni < 2; ++ni)
      bfr[ni] = load_b_frag(&Bs[(wn * 32 + ni * 16 + l16) * LDA + lh * 16]);
#pragma unroll
    for (int mi = 0; mi < 4; ++mi)
#pragma unroll
      for (int ni = 0; ni < 2; ++ni)
        acc[mi][ni] = wmma_bf16(afr[mi], bfr[ni], acc[mi][ni]);
  }

  // ---- store C (row = mbase + lh*8 + rr, col = nbase + l16) ----
#pragma unroll
  for (int mi = 0; mi < 4; ++mi) {
#pragma unroll
    for (int ni = 0; ni < 2; ++ni) {
      const int n     = bn * 128 + wn * 32 + ni * 16 + l16;
      const int mbase = bm * 128 + wm * 64 + mi * 16 + lh * 8;
#pragma unroll
      for (int rr = 0; rr < 8; ++rr) {
        const int m = mbase + rr;
        const float val = acc[mi][ni][rr];
        size_t idx;
        if constexpr (HEAD_LAYOUT) {
          const int b  = m >> 11;        // / SEQ
          const int s  = m & 2047;
          const int h  = n >> 7;         // / DK
          const int dk = n & 127;
          idx = (((size_t)(b * NHEADS + h)) * SEQ + s) * DK + dk;
        } else {
          idx = (size_t)m * N + n;
        }
        if constexpr (sizeof(OT) == 2) C[idx] = (OT)val;
        else                            C[idx] = val;
      }
    }
  }
}

// ---------------------------------------------------------------------------
// Flash attention. One block = 128 query rows of one (b,h); 8 waves x 16 rows.
// K tiles stream through LDS via async-to-LDS; online softmax; O in f32.
// ---------------------------------------------------------------------------
__global__ __launch_bounds__(256) void flash_attn(
    const __bf16* __restrict__ Qh, const __bf16* __restrict__ Kh,
    const __bf16* __restrict__ Vh, __bf16* __restrict__ attn) {
  constexpr int LDV = 40;  // Vst row pad
  __shared__ __bf16 Ks[32][DK];        // [key][dk]   8 KB
  __shared__ __bf16 Vst[DK][LDV];      // [dk][key]  10 KB (transposed)
  __shared__ __bf16 Ps[8][16][32];     // per-wave P  8 KB

  const int tid  = threadIdx.x;
  const int lane = tid & 31;
  const int wave = tid >> 5;
  const int l16  = lane & 15;
  const int lh   = lane >> 4;

  const int bh = blockIdx.y;           // b*16 + h
  const int b  = bh >> 4;
  const int h  = bh & 15;
  const int qbase = blockIdx.x * 128 + wave * 16;

  // ---- Q fragments for this wave's 16 rows, 4 K-chunks of 32 ----
  v16bf aq[4];
  {
    const __bf16* qrow = Qh + ((size_t)bh * SEQ + qbase + l16) * DK;
#pragma unroll
    for (int kc = 0; kc < 4; ++kc) aq[kc] = load_a_frag(qrow + kc * 32, lh);
  }

  float m_i[8], l_i[8];
#pragma unroll
  for (int rr = 0; rr < 8; ++rr) { m_i[rr] = -1e30f; l_i[rr] = 0.0f; }
  v8f o[8];
#pragma unroll
  for (int ni = 0; ni < 8; ++ni) o[ni] = (v8f)0.0f;

  const float scale = 0.08838834764831845f;  // 1/sqrt(128)

  const int key = tid >> 3;            // 0..31  (staging assignment)
  const int cb  = (tid & 7) * 16;      // 0..112

  for (int kt = 0; kt < SEQ; kt += 32) {
    __syncthreads();
    // ---- K tile: async global->LDS (no VGPR round trip) ----
    async_copy_32B(lds_off32(&Ks[key][cb]),
                   Kh + ((size_t)bh * SEQ + kt + key) * DK + cb);
    // ---- V tile: load + transpose into LDS ----
    {
      const __bf16* vsrc = Vh + ((size_t)bh * SEQ + kt + key) * DK + cb;
#pragma unroll
      for (int j = 0; j < 16; ++j) Vst[cb + j][key] = vsrc[j];
    }
    // ---- prefetch next key tile ----
    if (kt + 32 < SEQ) {
      __builtin_prefetch(Kh + ((size_t)bh * SEQ + kt + 32 + key) * DK + cb);
      __builtin_prefetch(Vh + ((size_t)bh * SEQ + kt + 32 + key) * DK + cb);
    }
    wait_async0();
    __syncthreads();

    // ---- S = Q K^T for 32 keys (two 16x16 tiles) ----
    v8f s0 = (v8f)0.0f, s1 = (v8f)0.0f;
#pragma unroll
    for (int kc = 0; kc < 4; ++kc) {
      v16bf bk0 = load_b_frag(&Ks[l16][kc * 32 + lh * 16]);
      v16bf bk1 = load_b_frag(&Ks[16 + l16][kc * 32 + lh * 16]);
      s0 = wmma_bf16(aq[kc], bk0, s0);
      s1 = wmma_bf16(aq[kc], bk1, s1);
    }

    // ---- online softmax (row lives across 16 lanes of a half-wave) ----
    float p0[8], p1[8], alpha[8];
#pragma unroll
    for (int rr = 0; rr < 8; ++rr) {
      float v0 = s0[rr] * scale, v1 = s1[rr] * scale;
      float mx = fmaxf(v0, v1);
#pragma unroll
      for (int off = 1; off < 16; off <<= 1)
        mx = fmaxf(mx, __shfl_xor(mx, off, 16));
      const float mnew = fmaxf(m_i[rr], mx);
      alpha[rr] = __expf(m_i[rr] - mnew);
      m_i[rr]   = mnew;
      p0[rr] = __expf(v0 - mnew);
      p1[rr] = __expf(v1 - mnew);
      float rs = p0[rr] + p1[rr];
#pragma unroll
      for (int off = 1; off < 16; off <<= 1)
        rs += __shfl_xor(rs, off, 16);
      l_i[rr] = l_i[rr] * alpha[rr] + rs;
    }
#pragma unroll
    for (int ni = 0; ni < 8; ++ni)
#pragma unroll
      for (int rr = 0; rr < 8; ++rr) o[ni][rr] *= alpha[rr];

    // ---- C-layout -> A-layout transpose of P through wave-private LDS ----
#pragma unroll
    for (int rr = 0; rr < 8; ++rr) {
      Ps[wave][lh * 8 + rr][l16]      = (__bf16)p0[rr];
      Ps[wave][lh * 8 + rr][16 + l16] = (__bf16)p1[rr];
    }
    asm volatile("s_wait_dscnt 0" ::: "memory");  // wave-private store->load
    v16bf ap = load_a_frag(&Ps[wave][l16][0], lh);

    // ---- O += P V ----
#pragma unroll
    for (int ni = 0; ni < 8; ++ni) {
      v16bf bv = load_b_frag(&Vst[ni * 16 + l16][lh * 16]);
      o[ni] = wmma_bf16(ap, bv, o[ni]);
    }
  }

  // ---- normalize and write attn in [B,S,D] bf16 ----
  float linv[8];
#pragma unroll
  for (int rr = 0; rr < 8; ++rr) linv[rr] = 1.0f / l_i[rr];
#pragma unroll
  for (int ni = 0; ni < 8; ++ni) {
#pragma unroll
    for (int rr = 0; rr < 8; ++rr) {
      const int s = qbase + lh * 8 + rr;
      const int d = h * DK + ni * 16 + l16;
      attn[((size_t)b * SEQ + s) * DMODEL + d] = (__bf16)(o[ni][rr] * linv[rr]);
    }
  }
}

// ---------------------------------------------------------------------------
extern "C" void kernel_launch(void* const* d_in, const int* in_sizes, int n_in,
                              void* d_out, int out_size, void* d_ws, size_t ws_size,
                              hipStream_t stream) {
  (void)in_sizes; (void)n_in; (void)out_size; (void)ws_size;
  const float* q  = (const float*)d_in[0];
  const float* k  = (const float*)d_in[1];
  const float* v  = (const float*)d_in[2];
  const float* wq = (const float*)d_in[3];
  const float* wk = (const float*)d_in[4];
  const float* wv = (const float*)d_in[5];
  const float* wo = (const float*)d_in[6];
  float* out = (float*)d_out;

  const size_t per = (size_t)MROWS * DMODEL;  // 16M elems
  __bf16* Qh = (__bf16*)d_ws;
  __bf16* Kh = Qh + per;
  __bf16* Vh = Kh + per;
  __bf16* At = Vh + per;

  dim3 blk(256);
  dim3 gproj(DMODEL / 128, MROWS / 128);  // (16, 64)

  gemm_wmma<float, __bf16, true><<<gproj, blk, 0, stream>>>(q, wq, Qh, MROWS, DMODEL, DMODEL);
  gemm_wmma<float, __bf16, true><<<gproj, blk, 0, stream>>>(k, wk, Kh, MROWS, DMODEL, DMODEL);
  gemm_wmma<float, __bf16, true><<<gproj, blk, 0, stream>>>(v, wv, Vh, MROWS, DMODEL, DMODEL);

  flash_attn<<<dim3(SEQ / 128, BATCH * NHEADS), blk, 0, stream>>>(Qh, Kh, Vh, At);

  gemm_wmma<__bf16, float, false><<<gproj, blk, 0, stream>>>(At, wo, out, MROWS, DMODEL, DMODEL);
}
